// AttentionLayer_7361573945659
// MI455X (gfx1250) — matrix operand order repaired and checked by reading, not compiled, not run on hardware
//
#include <hip/hip_runtime.h>
#include <math.h>

typedef __attribute__((ext_vector_type(16))) _Float16 v16h;
typedef __attribute__((ext_vector_type(8)))  _Float16 v8h;
typedef __attribute__((ext_vector_type(4)))  _Float16 v4h;
typedef __attribute__((ext_vector_type(8)))  float    v8f;
typedef __attribute__((ext_vector_type(4)))  float    v4f;

#define WMMA_F16(a, b, c) \
    __builtin_amdgcn_wmma_f32_16x16x32_f16(false, (a), false, (b), (short)0, (c), false, false)

// ---- CDNA5 async global->LDS copy (ASYNCcnt) -------------------------------
__device__ __forceinline__ void async_copy_b128(void* lds, const void* gaddr) {
    const uint32_t l = (uint32_t)(uintptr_t)lds;           // LDS byte address (addr[31:0])
    const unsigned long long g = (unsigned long long)(uintptr_t)gaddr;
    asm volatile("global_load_async_to_lds_b128 %0, %1, off"
                 :: "v"(l), "v"(g) : "memory");
}
__device__ __forceinline__ void wait_async_zero() {
    asm volatile("s_wait_asynccnt 0x0" ::: "memory");
}

// ---------------------------------------------------------------------------
// Kernel 0a: f32 -> f16 bulk convert (values)
// ---------------------------------------------------------------------------
__global__ __launch_bounds__(256) void cvt_f16_kernel(
    const float* __restrict__ in, _Float16* __restrict__ out, int n4)
{
    int i = blockIdx.x * 256 + threadIdx.x;
    const int stride = gridDim.x * 256;
    for (; i < n4; i += stride) {
        const v4f x = *(const v4f*)(in + (size_t)i * 4);
        v4h y;
        #pragma unroll
        for (int j = 0; j < 4; ++j) y[j] = (_Float16)x[j];
        *(v4h*)(out + (size_t)i * 4) = y;
    }
}

// ---------------------------------------------------------------------------
// Kernel 0b: W2 (1024x1024 f32, [k][n]) -> W2T (f16, [n][k]) via LDS tile
// ---------------------------------------------------------------------------
__global__ __launch_bounds__(256) void w2t_kernel(
    const float* __restrict__ W2, _Float16* __restrict__ W2T)
{
    __shared__ _Float16 tile[32][33];
    const int k0 = blockIdx.x * 32, n0 = blockIdx.y * 32;
    const int tx = threadIdx.x & 31, ty = threadIdx.x >> 5;   // 8 row-groups
    #pragma unroll
    for (int r = ty; r < 32; r += 8)
        tile[r][tx] = (_Float16)W2[(size_t)(k0 + r) * 1024 + n0 + tx];
    __syncthreads();
    #pragma unroll
    for (int r = ty; r < 32; r += 8)
        W2T[(size_t)(n0 + r) * 1024 + k0 + tx] = tile[tx][r];
}

// ---------------------------------------------------------------------------
// Kernel 1: q_proj = query(32x1024) @ W1(1024x1024) + b1 -> qproj (f32)
// One wave per 16x16 tile; tiny, direct-gather f32->f16 is fine here.
// ---------------------------------------------------------------------------
__global__ __launch_bounds__(32) void qproj_kernel(
    const float* __restrict__ query, const float* __restrict__ W1,
    const float* __restrict__ b1, float* __restrict__ qproj)
{
    const int nt    = blockIdx.x;        // 0..63
    const int mt    = blockIdx.y;        // 0..1
    const int lane  = threadIdx.x;
    const int mloc  = lane & 15;
    const int hi    = lane >> 4;
    const int khalf = hi * 8;
    const int kgrp  = hi * 16;
    const int col   = nt * 16 + mloc;
    const int row   = mt * 16 + mloc;

    v8f acc = {};
    for (int kb = 0; kb < 1024; kb += 32) {
        const float* ap = query + row * 1024 + kb + khalf;
        v16h a, b;
        #pragma unroll
        for (int j = 0; j < 8; ++j) {
            a[j]     = (_Float16)ap[j];
            a[j + 8] = (_Float16)ap[16 + j];
        }
        const float* bp = W1 + (size_t)(kb + khalf) * 1024 + col;
        #pragma unroll
        for (int j = 0; j < 16; ++j)
            b[j] = (_Float16)bp[(size_t)(j < 8 ? j : j + 8) * 1024];
        acc = WMMA_F16(a, b, acc);
    }

    const float bias = b1[col];
    #pragma unroll
    for (int r = 0; r < 8; ++r) {
        const int m = mt * 16 + r + 8 * hi;
        qproj[m * 1024 + col] = acc[r] + bias;
    }
}

// ---------------------------------------------------------------------------
// Kernel 2: scores[b,s] = V . tanh(qproj[b] + values[b,s,:]@W2 + b2) + bV
// 1024 blocks x 128 threads (4 waves), one batch's 64 rows per block.
// All operands pre-converted f16; staging via async b128 copies to LDS:
//   lds_a : 64x1024 f16 (128 KB, contiguous chunk of values16)
//   lds_b : double-buffered 16x1024 f16 panels (2 x 32 KB, contiguous in W2T)
// Each wave: one 16-row M-tile, 64 N-tiles x 32 fully-unrolled WMMAs.
// ---------------------------------------------------------------------------
__global__ __launch_bounds__(128) void scores_kernel(
    const _Float16* __restrict__ values16, const _Float16* __restrict__ W2T,
    const float* __restrict__ b2, const float* __restrict__ qproj,
    const float* __restrict__ Vw, const float* __restrict__ bV,
    float* __restrict__ scores)
{
    extern __shared__ _Float16 smem[];
    _Float16* lds_a  = smem;                    // [64][1024]
    _Float16* lds_b0 = smem + 64 * 1024;        // [16][1024] buffer 0
    _Float16* lds_b1 = lds_b0 + 16 * 1024;      // [16][1024] buffer 1

    const int tid  = threadIdx.x;
    const int lane = tid & 31;
    const int wave = tid >> 5;                  // 0..3
    const int blk  = blockIdx.x;
    const int b    = blk >> 5;
    const int s0   = (blk & 31) * 64;
    const size_t row0 = (size_t)(b * 2048 + s0);

    // Async-stage A (64 rows, 128 KB contiguous) and B panel 0 (32 KB contiguous)
    const _Float16* aG = values16 + row0 * 1024;
    for (int i = tid; i < 8192; i += 128)                 // 8192 x 16B chunks
        async_copy_b128(lds_a + i * 8, aG + i * 8);
    for (int i = tid; i < 2048; i += 128)                 // 2048 x 16B chunks
        async_copy_b128(lds_b0 + i * 8, W2T + i * 8);
    wait_async_zero();
    __syncthreads();

    const int mloc  = lane & 15;
    const int hi    = lane >> 4;
    const int khalf = hi * 8;
    const int kgrp  = hi * 16;
    const int arow  = wave * 16 + mloc;

    float part[8];
    #pragma unroll
    for (int r = 0; r < 8; ++r) part[r] = 0.f;

    for (int nt = 0; nt < 64; ++nt) {
        const _Float16* cur = (nt & 1) ? lds_b1 : lds_b0;
        _Float16*       nxt = (nt & 1) ? lds_b0 : lds_b1;

        // Overlap: async-stage next W2T panel while computing this one
        if (nt + 1 < 64) {
            const _Float16* g = W2T + (size_t)(nt + 1) * 16 * 1024;
            for (int i = tid; i < 2048; i += 128)
                async_copy_b128(nxt + i * 8, g + i * 8);
        }

        v8f acc = {};
        #pragma unroll
        for (int kk = 0; kk < 32; ++kk) {
            const int kb = kk * 32;
            const v8h alo = *(const v8h*)(lds_a + arow * 1024 + kb + khalf);
            const v8h ahi = *(const v8h*)(lds_a + arow * 1024 + kb + 16 + khalf);
            const v16h a = __builtin_shufflevector(alo, ahi,
                0, 1, 2, 3, 4, 5, 6, 7, 8, 9, 10, 11, 12, 13, 14, 15);
            const v8h blo = *(const v8h*)(cur + mloc * 1024 + kb + khalf);
            const v8h bhi = *(const v8h*)(cur + mloc * 1024 + kb + 16 + khalf);
            const v16h bb = __builtin_shufflevector(blo, bhi,
                0, 1, 2, 3, 4, 5, 6, 7, 8, 9, 10, 11, 12, 13, 14, 15);
            acc = WMMA_F16(a, bb, acc);
        }

        // Fused epilogue: tanh(c + qproj + b2) * V
        const int col = nt * 16 + mloc;
        const float qv = qproj[b * 1024 + col] + b2[col];
        const float vv = Vw[col];
        #pragma unroll
        for (int r = 0; r < 8; ++r)
            part[r] += tanhf(acc[r] + qv) * vv;

        wait_async_zero();    // next panel landed
        __syncthreads();
    }

    // Reduce across the 16-lane n-groups
    #pragma unroll
    for (int msk = 1; msk <= 8; msk <<= 1) {
        #pragma unroll
        for (int r = 0; r < 8; ++r)
            part[r] += __shfl_xor(part[r], msk, 32);
    }

    if (mloc == 0) {
        const float bv = bV[0];
        #pragma unroll
        for (int r = 0; r < 8; ++r) {
            const int s = s0 + wave * 16 + r + 8 * hi;
            scores[b * 2048 + s] = part[r] + bv;
        }
    }
}

// ---------------------------------------------------------------------------
// Kernel 3: softmax over S per batch, then context[b,d] = sum_s attn*values
// Reads the f16 copy of values (halves HBM traffic); f32 accumulation.
// ---------------------------------------------------------------------------
__global__ __launch_bounds__(256) void softmax_ctx_kernel(
    const _Float16* __restrict__ values16, const float* __restrict__ scores,
    float* __restrict__ out)
{
    __shared__ float attn[2048];
    __shared__ float red[256];
    const int b = blockIdx.x;
    const int t = threadIdx.x;

    float mx = -1e30f;
    for (int s = t; s < 2048; s += 256) {
        const float v = scores[b * 2048 + s];
        attn[s] = v;
        mx = fmaxf(mx, v);
    }
    red[t] = mx;
    __syncthreads();
    for (int off = 128; off > 0; off >>= 1) {
        if (t < off) red[t] = fmaxf(red[t], red[t + off]);
        __syncthreads();
    }
    mx = red[0];
    __syncthreads();

    float sum = 0.f;
    for (int s = t; s < 2048; s += 256) {
        const float e = __expf(attn[s] - mx);
        attn[s] = e;
        sum += e;
    }
    red[t] = sum;
    __syncthreads();
    for (int off = 128; off > 0; off >>= 1) {
        if (t < off) red[t] += red[t + off];
        __syncthreads();
    }
    const float inv = 1.f / red[0];
    __syncthreads();

    v4f acc = {};
    const _Float16* vb = values16 + (size_t)b * 2048 * 1024 + t * 4;
    for (int s = 0; s < 2048; ++s) {
        const float w = attn[s];
        const v4h x = *(const v4h*)(vb + (size_t)s * 1024);
        #pragma unroll
        for (int j = 0; j < 4; ++j) acc[j] += w * (float)x[j];
    }
    acc *= inv;
    *(v4f*)(out + b * 1024 + t * 4) = acc;
}

// ---------------------------------------------------------------------------
// Launch. Inputs: 0=values 1=query 2=W1 3=b1 4=W2 5=b2 6=V 7=bV (all f32)
// ws: values16 (32*2048*1024 f16) | W2T (1024*1024 f16) | qproj f32 | scores f32
// ---------------------------------------------------------------------------
extern "C" void kernel_launch(void* const* d_in, const int* in_sizes, int n_in,
                              void* d_out, int out_size, void* d_ws, size_t ws_size,
                              hipStream_t stream)
{
    const float* values = (const float*)d_in[0];
    const float* query  = (const float*)d_in[1];
    const float* W1     = (const float*)d_in[2];
    const float* b1     = (const float*)d_in[3];
    const float* W2     = (const float*)d_in[4];
    const float* b2     = (const float*)d_in[5];
    const float* Vw     = (const float*)d_in[6];
    const float* bV     = (const float*)d_in[7];
    float* out = (float*)d_out;

    _Float16* values16 = (_Float16*)d_ws;                        // 67108864 f16
    _Float16* W2T      = values16 + (size_t)32 * 2048 * 1024;    // 1048576 f16
    float*    qproj    = (float*)(W2T + 1024 * 1024);            // 32768 f32
    float*    scores   = qproj + 32 * 1024;                      // 65536 f32

    // Pre-convert (one pass over values: 256 MiB read, 128 MiB write)
    cvt_f16_kernel<<<4096, 256, 0, stream>>>(values, values16, 32 * 2048 * 1024 / 4);
    w2t_kernel<<<dim3(32, 32), 256, 0, stream>>>(W2, W2T);

    qproj_kernel<<<dim3(64, 2), 32, 0, stream>>>(query, W1, b1, qproj);

    const size_t lds2 = (size_t)(64 * 1024 + 2 * 16 * 1024) * sizeof(_Float16); // 192 KB
    scores_kernel<<<dim3(1024), 128, lds2, stream>>>(values16, W2T, b2, qproj, Vw, bV, scores);

    softmax_ctx_kernel<<<dim3(32), 256, 0, stream>>>(values16, scores, out);
}
